// CE_RVQ_18889266167841
// MI455X (gfx1250) — matrix operand order repaired and checked by reading, not compiled, not run on hardware
//
#include <hip/hip_runtime.h>
#include <hip/hip_bf16.h>
#include <math.h>

#define B_  16
#define D_  256
#define T_  4096
#define NQ_ 8
#define K_  1024
#define DC_ 128
#define BT_ (B_*T_)

typedef _Float16 half_t;
typedef __attribute__((ext_vector_type(16))) _Float16 v16h;
typedef __attribute__((ext_vector_type(8)))  float    v8f;
typedef int vint4 __attribute__((vector_size(16)));   // matches builtin's int4-vector param

#if defined(__AMDGCN__) && __has_builtin(__builtin_amdgcn_global_load_async_to_lds_b128)
#define HAVE_ASYNC_LDS 1
#else
#define HAVE_ASYNC_LDS 0
#endif

static __device__ __forceinline__ v8f wmma_f32_f16(v16h a, v16h b, v8f c) {
  // D = A(16x32 f16) * B(32x16 f16) + C(16x16 f32)
  return __builtin_amdgcn_wmma_f32_16x16x32_f16(
      /*neg_a=*/false, a, /*neg_b=*/false, b,
      /*c_mod=*/(short)0, c, /*reuse_a=*/false, /*reuse_b=*/false);
}

// ---- A-fragment (16x32, 16-bit) per ISA 7.12.2:
// lane holds row m = lane%16 (both halves), element j -> K = (j/8)*16 + (lane/16)*8 + (j%8)
static __device__ __forceinline__ v16h load_a_rowmajor(const half_t* base, int ld, int m0, int kbase, int lane) {
  int m = m0 + (lane & 15);
  int h = lane >> 4;
  const half_t* p = base + (size_t)m * ld + kbase;
  v16h a;
#pragma unroll
  for (int j = 0; j < 8; ++j) a[j]     = p[h * 8 + j];
#pragma unroll
  for (int j = 0; j < 8; ++j) a[8 + j] = p[16 + h * 8 + j];
  return a;
}

// A-fragment from residual: A[m][k] = res[b, kbase+k, t], m = b*T + t (stride T_ between k)
static __device__ __forceinline__ v16h load_a_residual(const float* res, int m0, int kbase, int lane) {
  int m = m0 + (lane & 15);
  int b = m >> 12;          // T_ = 4096
  int t = m & (T_ - 1);
  int h = lane >> 4;
  const float* p = res + (size_t)b * D_ * T_ + t;
  v16h a;
#pragma unroll
  for (int j = 0; j < 8; ++j) a[j]     = (_Float16)p[(size_t)(kbase + h * 8 + j) * T_];
#pragma unroll
  for (int j = 0; j < 8; ++j) a[8 + j] = (_Float16)p[(size_t)(kbase + 16 + h * 8 + j) * T_];
  return a;
}

// ---- B-fragment (32x16, 16-bit): lane holds column n_local = lane%16, element j -> K = (j/8)*16 + (lane/16)*8 + (j%8).
// Source layout: src[n][kdim] row-major with leading dim ld (per-lane load = two runs of 8 halves).
static __device__ __forceinline__ v16h load_b_rowT(const half_t* base, int ld, int n0, int kbase, int lane) {
  int n = n0 + (lane & 15);
  int h = lane >> 4;
  const half_t* p = base + (size_t)n * ld + kbase;
  v16h b;
#pragma unroll
  for (int j = 0; j < 8; ++j) { b[j] = p[h * 8 + j]; b[8 + j] = p[16 + h * 8 + j]; }
  return b;
}

// B-fragment from an LDS-resident 16 x DC_ chunk (rows are local 0..15).
static __device__ __forceinline__ v16h load_b_lds(const half_t* chunk, int kbase, int lane) {
  int n = lane & 15;
  int h = lane >> 4;
  const half_t* p = chunk + n * DC_ + kbase;
  v16h b;
#pragma unroll
  for (int j = 0; j < 8; ++j) { b[j] = p[h * 8 + j]; b[8 + j] = p[16 + h * 8 + j]; }
  return b;
}

// Cooperatively stage one 16 x DC_ f16 chunk (4 KB) into LDS: thread tid moves 16 bytes.
static __device__ __forceinline__ void stage_chunk(const half_t* __restrict__ src,
                                                   half_t* dst, int tid) {
#if HAVE_ASYNC_LDS
  __builtin_amdgcn_global_load_async_to_lds_b128(
      (__attribute__((address_space(1))) vint4*)(src + tid * 8),
      (__attribute__((address_space(3))) vint4*)(dst + tid * 8),
      0, 0);
#else
  ((vint4*)dst)[tid] = ((const vint4*)src)[tid];
#endif
}

static __device__ __forceinline__ void wait_async_deposits() {
#if HAVE_ASYNC_LDS
  __builtin_amdgcn_s_wait_asynccnt(0);
#endif
}

// ============================ kernels ============================

__global__ void prep_kernel(const float* __restrict__ diff, const float* __restrict__ Win,
                            const float* __restrict__ Wout, const float* __restrict__ embed,
                            float* res, half_t* win16, half_t* wout16, half_t* emb16,
                            float* e2, float* loss) {
  size_t i0 = (size_t)blockIdx.x * blockDim.x + threadIdx.x;
  size_t stride = (size_t)gridDim.x * blockDim.x;
  for (size_t i = i0; i < (size_t)B_ * D_ * T_; i += stride) res[i] = diff[i];
  for (size_t i = i0; i < (size_t)NQ_ * DC_ * D_; i += stride) win16[i]  = (_Float16)Win[i];
  for (size_t i = i0; i < (size_t)NQ_ * D_ * DC_; i += stride) wout16[i] = (_Float16)Wout[i];
  for (size_t i = i0; i < (size_t)NQ_ * K_ * DC_; i += stride) emb16[i]  = (_Float16)embed[i];
  for (size_t i = i0; i < (size_t)NQ_ * K_; i += stride) {
    const float* p = embed + i * DC_;
    float s = 0.f;
    for (int c = 0; c < DC_; ++c) s += p[c] * p[c];
    e2[i] = s;
  }
  if (i0 == 0) *loss = 0.f;
}

// xp[m][c] = sum_d res[b,d,t] * Win[c][d] + b_in[c]   (M=BT, N=DC, K=D)
__global__ void project_kernel(const float* __restrict__ res, const half_t* __restrict__ win16,
                               const float* __restrict__ b_in, half_t* __restrict__ xp16) {
  int wave = blockIdx.x * (blockDim.x >> 5) + (threadIdx.x >> 5);
  int lane = threadIdx.x & 31;
  int m0 = wave * 16;
  v16h afr[8];
#pragma unroll
  for (int kc = 0; kc < 8; ++kc) afr[kc] = load_a_residual(res, m0, kc * 32, lane);
  int h = lane >> 4, nl = lane & 15;
#pragma unroll 1
  for (int nc = 0; nc < DC_ / 16; ++nc) {
    v8f acc = {};
#pragma unroll
    for (int kc = 0; kc < 8; ++kc) {
      v16h bfr = load_b_rowT(win16, D_, nc * 16, kc * 32, lane);
      acc = wmma_f32_f16(afr[kc], bfr, acc);
    }
    int c = nc * 16 + nl;
    float bias = b_in[c];
#pragma unroll
    for (int r = 0; r < 8; ++r) {
      int m = m0 + h * 8 + r;
      xp16[(size_t)m * DC_ + c] = (_Float16)(acc[r] + bias);
    }
  }
}

__global__ void rownorm_kernel(const half_t* __restrict__ xp16, float* __restrict__ x2) {
  int m = blockIdx.x * blockDim.x + threadIdx.x;
  if (m >= BT_) return;
  const half_t* p = xp16 + (size_t)m * DC_;
  float s = 0.f;
#pragma unroll 4
  for (int c = 0; c < DC_; ++c) { float v = (float)p[c]; s += v * v; }
  x2[m] = s;
}

// sq[m][k] = x2[m] - 2*xp.e[k] + e2[k]; argmin over k; optional online logsumexp of -sq/DC.
// Block = 8 waves; embed chunks double-buffered in LDS via async global->LDS loads.
__global__ void dist_kernel(const half_t* __restrict__ xp16, const half_t* __restrict__ emb16,
                            const float* __restrict__ x2, const float* __restrict__ e2,
                            int* __restrict__ ind, float* __restrict__ lse, int do_lse) {
  __shared__ half_t btile[2][16 * DC_];   // 2 x 4 KB double buffer
  int tid  = threadIdx.x;
  int wave = tid >> 5;
  int lane = tid & 31;
  int m0 = (blockIdx.x * 8 + wave) * 16;

  v16h afr[4];
#pragma unroll
  for (int kc = 0; kc < 4; ++kc) afr[kc] = load_a_rowmajor(xp16, DC_, m0, kc * 32, lane);
  int h = lane >> 4, nl = lane & 15;
  float rx2[8];
#pragma unroll
  for (int r = 0; r < 8; ++r) rx2[r] = x2[m0 + h * 8 + r];
  float minv[8], mmax[8], ssum[8];
  int   mini[8];
#pragma unroll
  for (int r = 0; r < 8; ++r) { minv[r] = 3.4e38f; mini[r] = 0; mmax[r] = -3.4e38f; ssum[r] = 0.f; }

  stage_chunk(emb16, &btile[0][0], tid);   // prefill buffer 0 with chunk 0

#pragma unroll 1
  for (int nc = 0; nc < K_ / 16; ++nc) {
    int cur = nc & 1;
    wait_async_deposits();   // this wave's async deposits complete
    __syncthreads();         // all waves' deposits done; prev-iter reads of other buffer done
    if (nc + 1 < K_ / 16)
      stage_chunk(emb16 + (size_t)(nc + 1) * 16 * DC_, &btile[cur ^ 1][0], tid);

    v8f acc = {};
#pragma unroll
    for (int kc = 0; kc < 4; ++kc) {
      v16h bfr = load_b_lds(&btile[cur][0], kc * 32, lane);
      acc = wmma_f32_f16(afr[kc], bfr, acc);
    }
    int k = nc * 16 + nl;
    float ek = e2[k];
#pragma unroll
    for (int r = 0; r < 8; ++r) {
      float sq = rx2[r] - 2.f * acc[r] + ek;
      if (sq < minv[r]) { minv[r] = sq; mini[r] = k; }
      if (do_lse) {
        float lg = -sq * (1.0f / DC_);
        float mn = fmaxf(mmax[r], lg);
        ssum[r] = ssum[r] * expf(mmax[r] - mn) + expf(lg - mn);
        mmax[r] = mn;
      }
    }
  }
  // reduce across the 16 lanes of each half-wave (xor masks < 16 stay in-group)
#pragma unroll
  for (int off = 1; off < 16; off <<= 1) {
#pragma unroll
    for (int r = 0; r < 8; ++r) {
      float ov = __shfl_xor(minv[r], off, 32);
      int   oi = __shfl_xor(mini[r], off, 32);
      if (ov < minv[r] || (ov == minv[r] && oi < mini[r])) { minv[r] = ov; mini[r] = oi; }
      if (do_lse) {
        float om = __shfl_xor(mmax[r], off, 32);
        float os = __shfl_xor(ssum[r], off, 32);
        float mn = fmaxf(mmax[r], om);
        ssum[r] = ssum[r] * expf(mmax[r] - mn) + os * expf(om - mn);
        mmax[r] = mn;
      }
    }
  }
  if (nl == 0) {
#pragma unroll
    for (int r = 0; r < 8; ++r) {
      int m = m0 + h * 8 + r;
      ind[m] = mini[r];
      if (do_lse) lse[m] = mmax[r] + logf(ssum[r]);
    }
  }
}

__global__ void ce_kernel(const half_t* __restrict__ xp16, const half_t* __restrict__ emb16,
                          const float* __restrict__ x2, const float* __restrict__ e2,
                          const float* __restrict__ lse, const int* __restrict__ tgt,
                          int stage, float* __restrict__ loss) {
  int m = blockIdx.x * blockDim.x + threadIdx.x;
  if (m >= BT_) return;
  int b = m >> 12, t = m & (T_ - 1);
  int k = tgt[((size_t)b * NQ_ + stage) * T_ + t];
  const half_t* xr = xp16 + (size_t)m * DC_;
  const half_t* er = emb16 + (size_t)k * DC_;
  float dot = 0.f;
#pragma unroll 4
  for (int c = 0; c < DC_; ++c) dot += (float)xr[c] * (float)er[c];
  float sq = x2[m] - 2.f * dot + e2[k];
  float picked = -sq * (1.0f / DC_);
  atomicAdd(loss, (lse[m] - picked) * (1.0f / (4.0f * BT_)));
}

// residual[b,d,t] -= (embed[ind[m]] . Wout[d,:]) + b_out[d]   (M=BT, N=D, K=DC)
__global__ void update_kernel(float* __restrict__ res, const half_t* __restrict__ emb16,
                              const half_t* __restrict__ wout16, const float* __restrict__ b_out,
                              const int* __restrict__ ind) {
  int wave = blockIdx.x * (blockDim.x >> 5) + (threadIdx.x >> 5);
  int lane = threadIdx.x & 31;
  int m0 = wave * 16;
  int h = lane >> 4, nl = lane & 15;
  int row = ind[m0 + nl];                 // gather codebook row for A-row = lane%16
  const half_t* p = emb16 + (size_t)row * DC_;
  v16h afr[4];
#pragma unroll
  for (int kc = 0; kc < 4; ++kc) {
#pragma unroll
    for (int j = 0; j < 8; ++j) {
      afr[kc][j]     = p[kc * 32 + h * 8 + j];
      afr[kc][8 + j] = p[kc * 32 + 16 + h * 8 + j];
    }
  }
#pragma unroll 1
  for (int nc = 0; nc < D_ / 16; ++nc) {
    v8f acc = {};
#pragma unroll
    for (int kc = 0; kc < 4; ++kc) {
      v16h bfr = load_b_rowT(wout16, DC_, nc * 16, kc * 32, lane);
      acc = wmma_f32_f16(afr[kc], bfr, acc);
    }
    int d = nc * 16 + nl;
    float bo = b_out[d];
#pragma unroll
    for (int r = 0; r < 8; ++r) {
      int m = m0 + h * 8 + r;
      int b = m >> 12, t = m & (T_ - 1);
      size_t idx = (size_t)b * D_ * T_ + (size_t)d * T_ + t;
      res[idx] -= (acc[r] + bo);
    }
  }
}

__global__ void writeout_kernel(const float* __restrict__ loss, float* __restrict__ out) {
  if (threadIdx.x == 0 && blockIdx.x == 0) out[0] = loss[0];
}

// ============================ launch ============================

extern "C" void kernel_launch(void* const* d_in, const int* in_sizes, int n_in,
                              void* d_out, int out_size, void* d_ws, size_t ws_size,
                              hipStream_t stream) {
  const float* diff  = (const float*)d_in[0];
  const int*   tgt   = (const int*)  d_in[1];
  const float* Win   = (const float*)d_in[2];
  const float* b_in  = (const float*)d_in[3];
  const float* Wout  = (const float*)d_in[4];
  const float* b_out = (const float*)d_in[5];
  const float* embed = (const float*)d_in[6];

  char* ws = (char*)d_ws;
  float*  res    = (float*) ws; ws += sizeof(float)  * (size_t)B_ * D_ * T_;     // 64 MiB
  half_t* xp16   = (half_t*)ws; ws += sizeof(half_t) * (size_t)BT_ * DC_;        // 16 MiB
  half_t* win16  = (half_t*)ws; ws += sizeof(half_t) * (size_t)NQ_ * DC_ * D_;
  half_t* wout16 = (half_t*)ws; ws += sizeof(half_t) * (size_t)NQ_ * D_ * DC_;
  half_t* emb16  = (half_t*)ws; ws += sizeof(half_t) * (size_t)NQ_ * K_ * DC_;
  float*  x2     = (float*) ws; ws += sizeof(float)  * (size_t)BT_;
  float*  e2     = (float*) ws; ws += sizeof(float)  * (size_t)NQ_ * K_;
  float*  lse    = (float*) ws; ws += sizeof(float)  * (size_t)BT_;
  int*    indb   = (int*)   ws; ws += sizeof(int)    * (size_t)BT_;
  float*  loss   = (float*) ws; ws += sizeof(float);

  prep_kernel<<<2048, 256, 0, stream>>>(diff, Win, Wout, embed, res, win16, wout16, emb16, e2, loss);

  const int WAVES   = BT_ / 16;           // 4096 wmma tiles
  const int WBLOCKS = WAVES / 8;          // 8 waves (256 thr) per block -> 512 blocks
  for (int i = 0; i < NQ_; ++i) {
    project_kernel<<<WBLOCKS, 256, 0, stream>>>(res, win16 + (size_t)i * DC_ * D_,
                                                b_in + (size_t)i * DC_, xp16);
    rownorm_kernel<<<BT_ / 256, 256, 0, stream>>>(xp16, x2);
    dist_kernel<<<WBLOCKS, 256, 0, stream>>>(xp16, emb16 + (size_t)i * K_ * DC_,
                                             x2, e2 + (size_t)i * K_, indb, lse,
                                             (i < 4) ? 1 : 0);
    if (i < 4)
      ce_kernel<<<BT_ / 256, 256, 0, stream>>>(xp16, emb16 + (size_t)i * K_ * DC_,
                                               x2, e2 + (size_t)i * K_, lse, tgt, i, loss);
    update_kernel<<<WBLOCKS, 256, 0, stream>>>(res, emb16 + (size_t)i * K_ * DC_,
                                               wout16 + (size_t)i * D_ * DC_,
                                               b_out + (size_t)i * D_, indb);
  }
  writeout_kernel<<<1, 32, 0, stream>>>(loss, (float*)d_out);
}
